// MARSStage_44255343018518
// MI455X (gfx1250) — matrix operand order repaired and checked by reading, not compiled, not run on hardware
//
#include <hip/hip_runtime.h>
#include <hip/hip_bf16.h>
#include <math.h>

// ---------------------------------------------------------------------------
// MARS stage for MI455X (gfx1250, wave32).  All dense contractions run on the
// WMMA pipe (v_wmma_f32_16x16x32_f16, f32 accumulate).  All GEMMs are
// tile-exact (M%64==0, N%16==0, K%32==0; odd weights zero-padded in ws) so
// fragment loads are unconditional.  Each wave computes a 64x16 super-tile:
// one B fragment feeds 4 back-to-back WMMAs (4x B reuse, full matrix pipe).
// The selective-SSM scan is parallelized over B*D_INNER channels with one
// 16-lane half-wave per channel (lane = state index).
// ---------------------------------------------------------------------------

typedef __attribute__((ext_vector_type(16))) _Float16 v16h;
typedef __attribute__((ext_vector_type(8)))  float    v8f;

#define Mtok 3136   // 4 * 28 * 28
#define Ltok 784
#define Bimg 4
#define Cch  192
#define DI   384
#define DS   16
#define DR   24
#define LD2  64     // padded stride of the x_w projection output

__device__ __forceinline__ float gelu_f(float x) {
    return 0.5f * x * (1.0f + erff(x * 0.70710678118654752f));
}
__device__ __forceinline__ float softplus_f(float x) {
    return (x > 20.0f) ? x : log1pf(__expf(x));
}

// ---------------------------------------------------------------------------
// WMMA GEMM:  C[M,N] = act(A[M,K] * Wt[N,K]^T + bias) + res
// PRECONDITIONS (enforced by callers): M%64==0, N%16==0, K%32==0.
// One wave per 64x16 super-tile: 4 accumulators share one B fragment.
// act: 0=none, 1=gelu, 2=softplus.
// ---------------------------------------------------------------------------
__global__ __launch_bounds__(256)
void k_gemm_wmma(const float* __restrict__ A, const float* __restrict__ Wt,
                 float* C, const float* __restrict__ bias, const float* res,
                 int M, int N, int K, int lda, int ldc, int act)
{
    const int lane = threadIdx.x & 31;
    const int wave = threadIdx.x >> 5;
    const int NT = N >> 4;
    const int MS = M >> 6;                      // 64-row super-tiles
    const int tile = blockIdx.x * 8 + wave;     // wave-uniform -> EXEC stays full
    if (tile >= MS * NT) return;
    const int ms = tile / NT, nt = tile % NT;
    const int m0 = ms << 6, n0 = nt << 4;
    const int hi = lane >> 4;
    const int lr = lane & 15;
    const float* Ar[4];
#pragma unroll
    for (int i = 0; i < 4; ++i)
        Ar[i] = A + (size_t)(m0 + i * 16 + lr) * lda;   // A row of this lane
    const float* Wrow = Wt + (size_t)(n0 + lr) * K;     // weight row (out col)

    v8f acc[4] = {{}, {}, {}, {}};
    for (int kk = 0; kk < K; kk += 32) {
        // speculative prefetch of next weight slice (dropped silently if OOB)
        __builtin_prefetch(Wrow + kk + 32, 0, 1);
        v16h b;
#pragma unroll
        for (int e = 0; e < 16; ++e)            // ISA 16-bit B 32x16: k=e+(e&8)+8*hi
            b[e] = (_Float16)Wrow[kk + e + (e & 8) + (hi << 3)];
        v16h a[4];
#pragma unroll
        for (int i = 0; i < 4; ++i)
#pragma unroll
            for (int e = 0; e < 16; ++e) {      // ISA 16-bit A 16x32: k=e+(e&8)+8*hi
                a[i][e] = (_Float16)Ar[i][kk + e + (e & 8) + (hi << 3)];
            }
#pragma unroll
        for (int i = 0; i < 4; ++i)             // 4 independent WMMAs, shared B
            acc[i] = __builtin_amdgcn_wmma_f32_16x16x32_f16(
                false, a[i], false, b, (short)0, acc[i], false, false);
    }
    const int col = n0 + lr;
    float bv = bias ? bias[col] : 0.0f;
#pragma unroll
    for (int i = 0; i < 4; ++i) {
#pragma unroll
        for (int v = 0; v < 8; ++v) {
            int r = m0 + i * 16 + v + (hi << 3);    // ISA f32 C/D 16x16 layout
            float x = acc[i][v] + bv;
            if (act == 1)      x = gelu_f(x);
            else if (act == 2) x = softplus_f(x);
            if (res) x += res[(size_t)r * ldc + col];
            C[(size_t)r * ldc + col] = x;
        }
    }
}

// ---------------------------------------------------------------------------
// Downsample conv (stride-2 2x2, CIN=96 -> COUT=192) as im2col WMMA GEMM.
// K = 96*2*2 = 384, M = 3136, N = 192 (tile-exact, unconditional loads).
// ---------------------------------------------------------------------------
__global__ __launch_bounds__(256)
void k_ds_conv_wmma(const float* __restrict__ X, const float* __restrict__ Wt,
                    float* __restrict__ C)
{
    const int lane = threadIdx.x & 31;
    const int wave = threadIdx.x >> 5;
    const int NT = 12;                         // 192/16
    const int tile = blockIdx.x * 8 + wave;
    if (tile >= 196 * NT) return;
    const int mt = tile / NT, nt = tile % NT;
    const int m0 = mt << 4, n0 = nt << 4;
    const int hi = lane >> 4;
    const int lr = lane & 15;
    const int row = m0 + lr;
    const int col = n0 + lr;
    const int b   = row / Ltok;
    const int rr  = row % Ltok;
    const int oh  = rr / 28, ow = rr % 28;
    // k-invariant part of the input address for this output pixel
    const float* Xp = X + ((size_t)b * 96) * 56 * 56 + (size_t)(oh * 2) * 56 + (ow * 2);
    const float* Wrow = Wt + (size_t)col * 384;

    v8f acc = {};
    for (int kk = 0; kk < 384; kk += 32) {
        v16h a, bfr;
#pragma unroll
        for (int e = 0; e < 16; ++e) {
            int ka = kk + e + (e & 8) + (hi << 3);
            int ci = ka >> 2, kh = (ka >> 1) & 1, kw = ka & 1;
            a[e]   = (_Float16)Xp[((size_t)ci * 56 + kh) * 56 + kw];
            bfr[e] = (_Float16)Wrow[kk + e + (e & 8) + (hi << 3)];
        }
        acc = __builtin_amdgcn_wmma_f32_16x16x32_f16(
            false, a, false, bfr, (short)0, acc, false, false);
    }
#pragma unroll
    for (int v = 0; v < 8; ++v) {
        int r = m0 + v + (hi << 3);
        C[(size_t)r * Cch + col] = acc[v];
    }
}

// ---------------------------------------------------------------------------
// Zero-padded 2D copy (for odd-shaped weights -> tile-exact scratch copies).
// ---------------------------------------------------------------------------
__global__ __launch_bounds__(256)
void k_pad2d(const float* __restrict__ src, float* __restrict__ dst,
             int rs, int cs_, int rd, int cd)
{
    int idx = blockIdx.x * 256 + threadIdx.x;
    if (idx >= rd * cd) return;
    int r = idx / cd, c = idx % cd;
    dst[idx] = (r < rs && c < cs_) ? src[r * cs_ + c] : 0.0f;
}

// ---------------------------------------------------------------------------
// GroupNorm stats (32 groups, biased variance).  One block per (b, g).
// stats[bg] = mean, stats[128+bg] = rstd.
// ---------------------------------------------------------------------------
__global__ __launch_bounds__(256)
void k_gn_stats(const float* __restrict__ T, float* __restrict__ stats, float eps)
{
    const int bg = blockIdx.x;
    const int b = bg >> 5, g = bg & 31;
    const int gc = Cch / 32;                  // 6
    const int cnt = Ltok * gc;                // 4704
    float s = 0.0f, sq = 0.0f;
    for (int i = threadIdx.x; i < cnt; i += 256) {
        int l = i / gc, c = g * gc + i % gc;
        float v = T[((size_t)(b * Ltok + l)) * Cch + c];
        s += v; sq += v * v;
    }
    __shared__ float ss[256], sg[256];
    ss[threadIdx.x] = s; sg[threadIdx.x] = sq;
    __syncthreads();
    for (int o = 128; o > 0; o >>= 1) {
        if (threadIdx.x < o) { ss[threadIdx.x] += ss[threadIdx.x + o];
                               sg[threadIdx.x] += sg[threadIdx.x + o]; }
        __syncthreads();
    }
    if (threadIdx.x == 0) {
        float mean = ss[0] / (float)cnt;
        float var  = sg[0] / (float)cnt - mean * mean;
        stats[bg] = mean;
        stats[128 + bg] = rsqrtf(var + eps);
    }
}

__global__ __launch_bounds__(256)
void k_gn_apply(float* X, const float* __restrict__ stats,
                const float* __restrict__ w, const float* __restrict__ bb)
{
    int idx = blockIdx.x * 256 + threadIdx.x;
    if (idx >= Mtok * Cch) return;
    int c = idx % Cch, t = idx / Cch, b = t / Ltok;
    int bg = (b << 5) + c / (Cch / 32);
    float v = (X[idx] - stats[bg]) * stats[128 + bg];
    X[idx] = v * w[c] + bb[c];
}

// X0 += gelu(GN(T))
__global__ __launch_bounds__(256)
void k_gn_gelu_add(float* X0, const float* __restrict__ T,
                   const float* __restrict__ stats,
                   const float* __restrict__ w, const float* __restrict__ bb)
{
    int idx = blockIdx.x * 256 + threadIdx.x;
    if (idx >= Mtok * Cch) return;
    int c = idx % Cch, t = idx / Cch, b = t / Ltok;
    int bg = (b << 5) + c / (Cch / 32);
    float v = (T[idx] - stats[bg]) * stats[128 + bg] * w[c] + bb[c];
    X0[idx] += gelu_f(v);
}

// ---------------------------------------------------------------------------
// Depthwise conv (ks in {3,5,7}) from token-major input into CAT buffer.
// ---------------------------------------------------------------------------
__global__ __launch_bounds__(256)
void k_dwconv(const float* __restrict__ X, const float* __restrict__ Wd,
              float* __restrict__ O, int ks, int pad, int off)
{
    int idx = blockIdx.x * 256 + threadIdx.x;
    if (idx >= Mtok * Cch) return;
    int c = idx % Cch, t = idx / Cch;
    int b = t / Ltok, r = t % Ltok;
    int h = r / 28, w = r % 28;
    const float* wc = Wd + (size_t)c * ks * ks;
    float s = 0.0f;
    for (int kh = 0; kh < ks; ++kh) {
        int ih = h + kh - pad;
        if (ih < 0 || ih >= 28) continue;
        for (int kw = 0; kw < ks; ++kw) {
            int iw = w + kw - pad;
            if (iw < 0 || iw >= 28) continue;
            s += X[((size_t)(b * Ltok + ih * 28 + iw)) * Cch + c] * wc[kh * ks + kw];
        }
    }
    O[(size_t)t * 576 + off + c] = s;
}

// ---------------------------------------------------------------------------
// SE: global average pool, tiny MLP + softmax(3).
// ---------------------------------------------------------------------------
__global__ __launch_bounds__(256)
void k_se_pool(const float* __restrict__ X, float* __restrict__ pooled)
{
    int idx = blockIdx.x * 256 + threadIdx.x;
    if (idx >= Bimg * Cch) return;
    int b = idx / Cch, c = idx % Cch;
    float s = 0.0f;
    for (int l = 0; l < Ltok; ++l) s += X[((size_t)(b * Ltok + l)) * Cch + c];
    pooled[idx] = s * (1.0f / (float)Ltok);
}

__global__ __launch_bounds__(256)
void k_se_mlp(const float* __restrict__ pooled,
              const float* __restrict__ w1, const float* __restrict__ b1,
              const float* __restrict__ w2, const float* __restrict__ b2,
              float* se)
{
    __shared__ float hid[Bimg][48];
    int tid = threadIdx.x;
    if (tid < Bimg * 48) {
        int b = tid / 48, i = tid % 48;
        float s = b1[i];
        for (int c = 0; c < Cch; ++c) s += pooled[b * Cch + c] * w1[i * Cch + c];
        hid[b][i] = fmaxf(s, 0.0f);
    }
    __syncthreads();
    if (tid < Bimg * 3) {
        int b = tid / 3, j = tid % 3;
        float s = b2[j];
        for (int i = 0; i < 48; ++i) s += hid[b][i] * w2[j * 48 + i];
        se[b * 3 + j] = s;
    }
    __syncthreads();
    if (tid < Bimg) {
        int b = tid;
        float a0 = se[b * 3], a1 = se[b * 3 + 1], a2 = se[b * 3 + 2];
        float mx = fmaxf(a0, fmaxf(a1, a2));
        float e0 = __expf(a0 - mx), e1 = __expf(a1 - mx), e2 = __expf(a2 - mx);
        float inv = 1.0f / (e0 + e1 + e2);
        se[b * 3] = e0 * inv; se[b * 3 + 1] = e1 * inv; se[b * 3 + 2] = e2 * inv;
    }
}

// PW += se[b,0]*s3 + se[b,1]*s5 + se[b,2]*s7
__global__ __launch_bounds__(256)
void k_fuse(float* PW, const float* __restrict__ CAT, const float* __restrict__ se)
{
    int idx = blockIdx.x * 256 + threadIdx.x;
    if (idx >= Mtok * Cch) return;
    int c = idx % Cch, t = idx / Cch, b = t / Ltok;
    const float* sb = se + b * 3;
    size_t base = (size_t)t * 576 + c;
    PW[idx] += sb[0] * CAT[base] + sb[1] * CAT[base + 192] + sb[2] * CAT[base + 384];
}

// ---------------------------------------------------------------------------
// LayerNorm over C=192, one wave32 per token, shuffle reduction.
// ---------------------------------------------------------------------------
__global__ __launch_bounds__(256)
void k_layernorm(const float* __restrict__ X, float* __restrict__ O,
                 const float* __restrict__ w, const float* __restrict__ bb)
{
    int lane = threadIdx.x & 31;
    int m = blockIdx.x * 8 + (threadIdx.x >> 5);
    const float* xr = X + (size_t)m * Cch;
    float v[6]; float s = 0.0f, sq = 0.0f;
#pragma unroll
    for (int i = 0; i < 6; ++i) {
        v[i] = xr[lane + 32 * i]; s += v[i]; sq += v[i] * v[i];
    }
    for (int o = 16; o > 0; o >>= 1) {
        s  += __shfl_xor(s,  o, 32);
        sq += __shfl_xor(sq, o, 32);
    }
    float mean = s * (1.0f / 192.0f);
    float var  = sq * (1.0f / 192.0f) - mean * mean;
    float r = rsqrtf(var + 1e-6f);
    float* orow = O + (size_t)m * Cch;
#pragma unroll
    for (int i = 0; i < 6; ++i) {
        int c = lane + 32 * i;
        orow[c] = (v[i] - mean) * r * w[c] + bb[c];
    }
}

// ---------------------------------------------------------------------------
// Selective-SSM scan.  One 16-lane group per (b, d) channel, lane = state n.
// Direction traversal by index map; accumulates softmax(dir_w)[dir] * y into Y.
//   dir 0: h      tok = t
//   dir 1: h-rev  tok = 783 - t
//   dir 2: v      tok = (t%28)*28 + t/28
//   dir 3: v-rev  u = 783-t; tok = (u%28)*28 + u/28
// ---------------------------------------------------------------------------
__global__ __launch_bounds__(256)
void k_scan(const float* __restrict__ DTb, const float* __restrict__ XZ2,
            const float* __restrict__ XZ,  const float* __restrict__ Alog,
            const float* __restrict__ Dp,  const float* __restrict__ dirw,
            float* Y, int dir, int first)
{
    int tid = blockIdx.x * 256 + threadIdx.x;
    int grp = tid >> 4;
    int n = tid & 15;
    if (grp >= Bimg * DI) return;
    int b = grp / DI, d = grp % DI;

    float w0 = dirw[0], w1 = dirw[1], w2 = dirw[2], w3 = dirw[3];
    float mx = fmaxf(fmaxf(w0, w1), fmaxf(w2, w3));
    float e0 = __expf(w0 - mx), e1 = __expf(w1 - mx),
          e2 = __expf(w2 - mx), e3 = __expf(w3 - mx);
    float wd = (dir == 0 ? e0 : dir == 1 ? e1 : dir == 2 ? e2 : e3)
               / (e0 + e1 + e2 + e3);

    float a_dn = -__expf(Alog[d * DS + n]);
    float Dv = Dp[d];
    float h = 0.0f;
    int base = b * Ltok;

    for (int t = 0; t < Ltok; ++t) {
        int u = (dir == 1 || dir == 3) ? (Ltok - 1 - t) : t;
        int tok = (dir >= 2) ? ((u % 28) * 28 + u / 28) : u;
        int m = base + tok;
        float dtv = DTb[(size_t)m * DI + d];
        float xv  = XZ [(size_t)m * 768 + d];
        float bs  = XZ2[(size_t)m * LD2 + DR + n];
        float cs  = XZ2[(size_t)m * LD2 + DR + DS + n];
        h = __expf(dtv * a_dn) * h + dtv * bs * xv;
        float p = h * cs;
        p += __shfl_xor(p, 8, 32); p += __shfl_xor(p, 4, 32);
        p += __shfl_xor(p, 2, 32); p += __shfl_xor(p, 1, 32);
        if (n == 0) {
            float yv = wd * (p + xv * Dv);
            float* yp = &Y[(size_t)m * DI + d];
            *yp = first ? yv : (*yp + yv);
        }
    }
}

// Y *= silu(z) with z = XZ[:, 384:768]
__global__ __launch_bounds__(256)
void k_silu_mul(float* Y, const float* __restrict__ XZ)
{
    int idx = blockIdx.x * 256 + threadIdx.x;
    if (idx >= Mtok * DI) return;
    int d = idx % DI, m = idx / DI;
    float z = XZ[(size_t)m * 768 + DI + d];
    Y[idx] *= z / (1.0f + __expf(-z));
}

// token-major [M,192] -> NCHW fp32 output
__global__ __launch_bounds__(256)
void k_to_nchw(const float* __restrict__ X, float* __restrict__ out)
{
    int idx = blockIdx.x * 256 + threadIdx.x;
    if (idx >= Mtok * Cch) return;
    int c = idx % Cch, m = idx / Cch;
    int b = m / Ltok, r = m % Ltok;
    out[((size_t)(b * Cch + c)) * Ltok + r] = X[idx];
}

// ---------------------------------------------------------------------------
static inline int gemm_grid(int M, int N) {
    int MS = M / 64, NT = N / 16;
    return (MS * NT + 7) / 8;
}

extern "C" void kernel_launch(void* const* d_in, const int* in_sizes, int n_in,
                              void* d_out, int out_size, void* d_ws, size_t ws_size,
                              hipStream_t stream)
{
    (void)in_sizes; (void)n_in; (void)out_size; (void)ws_size;
#define P(i) ((const float*)d_in[(i)])
    const float* x = P(0);

    float* ws = (float*)d_ws;
    size_t o = 0;
    float* X0    = ws + o; o += (size_t)Mtok * Cch;   // current activation
    float* CAT   = ws + o; o += (size_t)Mtok * 576;   // [s3|s5|s7]
    float* PW    = ws + o; o += (size_t)Mtok * Cch;   // pw out / GN temp
    float* XN    = ws + o; o += (size_t)Mtok * Cch;   // LN output
    float* XZ    = ws + o; o += (size_t)Mtok * 768;   // in-proj (x_in | z)
    float* XZ2   = ws + o; o += (size_t)Mtok * LD2;   // x_w proj [M,64] (padded)
    float* DTb   = ws + o; o += (size_t)Mtok * DI;    // softplus(dt)
    float* Y     = ws + o; o += (size_t)Mtok * DI;    // scan accumulator
    float* H1    = ws + o; o += (size_t)Mtok * 768;   // FFN hidden
    float* XW64  = ws + o; o += (size_t)64 * 384;     // x_w zero-padded [64,384]
    float* DTW32 = ws + o; o += (size_t)DI * 32;      // dt_w zero-padded [384,32]
    float* STATS  = ws + o; o += 256;                 // GN mean | rstd
    float* POOLED = ws + o; o += 1024;
    float* SE     = ws + o; o += 16;

    const int EW = (Mtok * Cch + 255) / 256;          // elementwise grid

    // ---- Stage 1: downsample conv (WMMA) + GroupNorm -----------------------
    k_ds_conv_wmma<<<(196 * 12 + 7) / 8, 256, 0, stream>>>(x, P(1), X0);
    k_gn_stats<<<128, 256, 0, stream>>>(X0, STATS, 1e-5f);
    k_gn_apply<<<EW, 256, 0, stream>>>(X0, STATS, P(2), P(3));

    // ---- Stage 2: multi-scale depthwise + SE + 1x1 fusion ------------------
    k_dwconv<<<EW, 256, 0, stream>>>(X0, P(4), CAT, 3, 1, 0);
    k_dwconv<<<EW, 256, 0, stream>>>(X0, P(5), CAT, 5, 2, 192);
    k_dwconv<<<EW, 256, 0, stream>>>(X0, P(6), CAT, 7, 3, 384);
    k_se_pool<<<(Bimg * Cch + 255) / 256, 256, 0, stream>>>(X0, POOLED);
    k_se_mlp<<<1, 256, 0, stream>>>(POOLED, P(10), P(11), P(12), P(13), SE);
    k_gemm_wmma<<<gemm_grid(Mtok, Cch), 256, 0, stream>>>(
        CAT, P(7), PW, nullptr, nullptr, Mtok, Cch, 576, 576, Cch, 0);
    k_fuse<<<EW, 256, 0, stream>>>(PW, CAT, SE);
    k_gn_stats<<<128, 256, 0, stream>>>(PW, STATS, 1e-5f);
    k_gn_gelu_add<<<EW, 256, 0, stream>>>(X0, PW, STATS, P(8), P(9));

    // ---- Stage 3: VSS blocks ----------------------------------------------
    for (int blk = 0; blk < 2; ++blk) {
        const int p = 14 + blk * 31;
        // LN1 + in-proj
        k_layernorm<<<Mtok / 8, 256, 0, stream>>>(X0, XN, P(p + 0), P(p + 1));
        k_gemm_wmma<<<gemm_grid(Mtok, 768), 256, 0, stream>>>(
            XN, P(p + 2), XZ, nullptr, nullptr, Mtok, 768, Cch, Cch, 768, 0);
        // 4-direction selective SSM
        for (int dir = 0; dir < 4; ++dir) {
            const int sp = p + 4 + 5 * dir;   // dt_w, dt_b, x_w, A_log, D
            // pad odd-shaped weights so the GEMMs are tile-exact
            k_pad2d<<<(64 * 384 + 255) / 256, 256, 0, stream>>>(
                P(sp + 2), XW64, 56, 384, 64, 384);
            k_pad2d<<<(DI * 32 + 255) / 256, 256, 0, stream>>>(
                P(sp + 0), DTW32, DI, DR, DI, 32);
            // xz2 = x_in @ x_w^T   (N padded 56->64, garbage cols 56..63 unused)
            k_gemm_wmma<<<gemm_grid(Mtok, LD2), 256, 0, stream>>>(
                XZ, XW64, XZ2, nullptr, nullptr, Mtok, LD2, DI, 768, LD2, 0);
            // dt = softplus(dt_raw @ dt_w^T + dt_b)  (K padded 24->32; padded
            // weight cols are zero so the b_ssm columns of XZ2 contribute 0)
            k_gemm_wmma<<<gemm_grid(Mtok, DI), 256, 0, stream>>>(
                XZ2, DTW32, DTb, P(sp + 1), nullptr, Mtok, DI, 32, LD2, DI, 2);
            k_scan<<<(Bimg * DI * DS + 255) / 256, 256, 0, stream>>>(
                DTb, XZ2, XZ, P(sp + 3), P(sp + 4), P(p + 24), Y, dir, dir == 0 ? 1 : 0);
        }
        // gate, out-proj (+residual)
        k_silu_mul<<<(Mtok * DI + 255) / 256, 256, 0, stream>>>(Y, XZ);
        k_gemm_wmma<<<gemm_grid(Mtok, Cch), 256, 0, stream>>>(
            Y, P(p + 3), X0, nullptr, X0, Mtok, Cch, DI, DI, Cch, 0);
        // LN2 + FFN (+residual)
        k_layernorm<<<Mtok / 8, 256, 0, stream>>>(X0, XN, P(p + 25), P(p + 26));
        k_gemm_wmma<<<gemm_grid(Mtok, 768), 256, 0, stream>>>(
            XN, P(p + 27), H1, P(p + 28), nullptr, Mtok, 768, Cch, Cch, 768, 1);
        k_gemm_wmma<<<gemm_grid(Mtok, Cch), 256, 0, stream>>>(
            H1, P(p + 29), X0, P(p + 30), X0, Mtok, Cch, 768, 768, Cch, 0);
    }

    // ---- Output: NHWC-flat -> NCHW ----------------------------------------
    k_to_nchw<<<EW, 256, 0, stream>>>(X0, (float*)d_out);
#undef P
}
